// ShiftImputationModel_40200893890565
// MI455X (gfx1250) — matrix operand order repaired and checked by reading, not compiled, not run on hardware
//
#include <hip/hip_runtime.h>
#include <hip/hip_bf16.h>
#include <math.h>

// Problem constants (from reference)
#define BB    8192
#define KNB   64
#define SS    6
#define QDIM  512
#define HIDD  192
#define NHEAD 4
#define HDIM  48
#define NTAB  21

typedef __attribute__((ext_vector_type(16))) __bf16 v16bf;
typedef __attribute__((ext_vector_type(8)))  float  v8f;

static __device__ __forceinline__ __bf16 f2bf(float f) {
  unsigned u = __builtin_bit_cast(unsigned, f);
  unsigned r = u + 0x7fffu + ((u >> 16) & 1u);   // round-to-nearest-even
  unsigned short h = (unsigned short)(r >> 16);
  return __builtin_bit_cast(__bf16, h);
}
static __device__ __forceinline__ float bf2f(__bf16 b) {
  unsigned short h = __builtin_bit_cast(unsigned short, b);
  unsigned u = ((unsigned)h) << 16;
  return __builtin_bit_cast(float, u);
}
static __device__ __forceinline__ float gelu_f(float x) {
  return 0.5f * x * (1.0f + erff(x * 0.70710678118654752f));
}
static __device__ __forceinline__ float sigm(float x) {
  return 1.0f / (1.0f + expf(-x));
}

// ---- WMMA fragment loaders (CDNA5 16-bit layouts, ISA 7.12.2) --------------
// A (16x32, row-major [M][lda]): lane L<16 -> row L, K = {k0..k0+7, k0+16..k0+23}
//                                lane L>=16 -> row L-16, K = {k0+8..15, k0+24..31}
static __device__ __forceinline__ v16bf load_a_frag(const __bf16* base, int lda,
                                                    int row0, int k0, int lane) {
  int r = lane & 15, half = lane >> 4;
  const __bf16* p = base + (row0 + r) * lda + k0 + half * 8;
  v16bf a;
#pragma unroll
  for (int i = 0; i < 8; ++i) a[i] = p[i];
#pragma unroll
  for (int i = 0; i < 8; ++i) a[8 + i] = p[16 + i];
  return a;
}
// B (32x16) staged transposed in LDS as [N][ldb] (K contiguous):
// lane L<16 -> col L, K = {k0..k0+7, k0+16..k0+23} ; lane L>=16 -> col L-16, K = {k0+8..15, k0+24..31}
static __device__ __forceinline__ v16bf load_b_frag(const __bf16* base, int ldb,
                                                    int n0, int k0, int lane) {
  int r = lane & 15, half = lane >> 4;
  const __bf16* p = base + (n0 + r) * ldb + k0 + half * 8;
  v16bf b;
#pragma unroll
  for (int i = 0; i < 8; ++i) b[i] = p[i];
#pragma unroll
  for (int i = 0; i < 8; ++i) b[8 + i] = p[16 + i];
  return b;
}

// ============================================================================
// Kernel 1: q = gelu(qe@Wq1+bq1), tq = gelu(qe@Wtq+btq) for all B rows (WMMA)
// Tile: M=32, combined N=384 (q|tq), K=512 streamed in 64-chunks via LDS.
// ============================================================================
__global__ __launch_bounds__(256) void qtq_gemm_kernel(
    const float* __restrict__ qe,
    const float* __restrict__ Wq1, const float* __restrict__ bq1,
    const float* __restrict__ Wtq, const float* __restrict__ btq,
    float* __restrict__ qout, float* __restrict__ tqout) {
  const int tid  = threadIdx.x;
  const int lane = tid & 31;
  const int wave = tid >> 5;
  const int m0   = blockIdx.x * 32;

  __shared__ __bf16 As[32 * 72];     //  4.6 KB
  __shared__ __bf16 Ws[384 * 72];    // 55.3 KB (transposed [N][K])

  v8f acc[6] = {};

  for (int kc = 0; kc < QDIM; kc += 64) {
    for (int idx = tid; idx < 32 * 64; idx += 256) {
      int m = idx >> 6, kk = idx & 63;
      As[m * 72 + kk] = f2bf(qe[(size_t)(m0 + m) * QDIM + kc + kk]);
    }
    for (int idx = tid; idx < 384 * 64; idx += 256) {
      int n = idx % 384, kk = idx / 384;
      float w = (n < HIDD) ? Wq1[(size_t)(kc + kk) * HIDD + n]
                           : Wtq[(size_t)(kc + kk) * HIDD + (n - HIDD)];
      Ws[n * 72 + kk] = f2bf(w);
    }
    __syncthreads();
#pragma unroll
    for (int i = 0; i < 6; ++i) {
      int t48 = wave * 6 + i;
      int mi = t48 & 1, nj = t48 >> 1;           // 2 M-tiles x 24 N-tiles
#pragma unroll
      for (int ks = 0; ks < 64; ks += 32) {
        v16bf a = load_a_frag(As, 72, mi * 16, ks, lane);
        v16bf b = load_b_frag(Ws, 72, nj * 16, ks, lane);
        acc[i] = __builtin_amdgcn_wmma_f32_16x16x32_bf16(
            false, a, false, b, (short)0, acc[i], false, false);
      }
    }
    __syncthreads();
  }

  const int r = lane & 15, half = lane >> 4;
#pragma unroll
  for (int i = 0; i < 6; ++i) {
    int t48 = wave * 6 + i;
    int mi = t48 & 1, nj = t48 >> 1;
    int n = nj * 16 + r;
    float bias = (n < HIDD) ? bq1[n] : btq[n - HIDD];
    float* dst = (n < HIDD) ? qout : tqout;
    int col = (n < HIDD) ? n : n - HIDD;
#pragma unroll
    for (int ii = 0; ii < 8; ++ii) {
      int m = m0 + mi * 16 + half * 8 + ii;
      dst[(size_t)m * HIDD + col] = gelu_f(acc[i][ii] + bias);
    }
  }
}

// ============================================================================
// Kernel 2: fully fused per-batch-row model. 1 block (8 waves) per row b.
// ============================================================================
struct FusedArgs {
  const float *qe; const int *qrc; const float *qos; const float *qsm;
  const float *rs; const unsigned char *rsm; const int *rrc; const float *rd;
  const unsigned char *rv; const float *rctab;
  const float *emb16, *Wn, *bn, *Wqa, *bqa, *Wka, *Wva, *bva;
  const float *Ws1, *bs1, *Ws2, *bs2, *Wc1, *bc1, *Wc2, *bc2;
  const float *Wt1, *bt1, *Wt2, *bt2, *emb32, *Wtc, *btc;
  const float *Wp1, *bp1, *Wp2, *bp2, *scale, *fb_shift, *fb_ctx;
  const float *qrow, *tqrow;                    // from kernel 1 (workspace)
  float *out_ctx, *out_trans, *out_trust;
};

__global__ __launch_bounds__(256) void fused_kernel(FusedArgs A) {
  const int b    = blockIdx.x;
  const int tid  = threadIdx.x;
  const int lane = tid & 31;
  const int wave = tid >> 5;
  const int r16  = lane & 15, half = lane >> 4;

  // ---- persistent LDS (~33 KB) ----
  __shared__ __bf16 sh_kv[KNB * 200];            // 25.6 KB  kv (bf16)
  __shared__ float sh_shifts[KNB][8];            // corrected shifts
  __shared__ float sh_vmf[KNB][8];               // rv & rsm
  __shared__ float sh_same[KNB], sh_rdv[KNB], sh_keep[KNB], sh_rvf[KNB];
  __shared__ int   sh_rrc[KNB];
  __shared__ float sh_scores[NHEAD][KNB];        // scores -> attn (in place)
  __shared__ float sh_sb[KNB][NHEAD];            // re-rank bias accumulator
  __shared__ float sh_qos[8], sh_qsm[8], sh_flags[4];

  // ---- overlaid LDS (18.4 KB): WMMA staging, then phase-B vectors ----
  __shared__ union {
    struct { __bf16 nib[KNB * 36]; __bf16 wnT[HIDD * 36]; } a;
    struct {
      float qv[HIDD], tqv[HIDD], Qv[HIDD];
      float Aproj[NHEAD][HIDD];
      float u[NHEAD][HIDD];
      float ao[HIDD], c1[HIDD], t1[HIDD], tcv[HIDD];
      float pss[SS][4], gst[4], swb[8], tracc[8];
    } bb;
  } ov;

  const int qrc_raw = A.qrc[b];
  const int qclip = min(max(qrc_raw, 0), NTAB - 1);

  // ---- stage 0: per-neighbor scalars, corrected shifts, masks -------------
  if (tid < SS) { sh_qos[tid] = A.qos[b * SS + tid]; sh_qsm[tid] = A.qsm[b * SS + tid]; }
  if (tid < KNB) {
    int k = tid;
    sh_rdv[k] = A.rd[(size_t)b * KNB + k];
    int rck = A.rrc[(size_t)b * KNB + k];
    sh_rrc[k]  = rck;
    sh_same[k] = (rck == qrc_raw) ? 1.0f : 0.0f;
    sh_rvf[k]  = (A.rv[(size_t)b * KNB + k] != 0) ? 1.0f : 0.0f;
  }
  sh_sb[tid >> 2][tid & 3] = 0.0f;               // zero 64x4
  for (int idx = tid; idx < KNB * SS; idx += 256) {
    int k = idx / SS, s = idx - k * SS;
    float rsv = A.rs[((size_t)b * KNB + k) * SS + s];
    int rci = min(max(A.rrc[(size_t)b * KNB + k], 0), NTAB - 1);
    float rcq = A.rctab[qclip * SS + s];
    float rcr = A.rctab[rci * SS + s];
    bool msk = (A.rsm[((size_t)b * KNB + k) * SS + s] != 0);
    bool ok  = (rcq == rcq) && (rcr == rcr) && msk;   // NaN checks
    sh_shifts[k][s] = ok ? (rcq + rsv - rcr) : rsv;
    bool rvk = (A.rv[(size_t)b * KNB + k] != 0);
    sh_vmf[k][s] = (rvk && msk) ? 1.0f : 0.0f;
  }
  __syncthreads();

  // ---- stage 1: any_valid + build WMMA operands (ni 64x32, WnT 192x32) ----
  if (tid == 0) {
    float a = 0.0f;
    for (int k = 0; k < KNB; ++k) a += sh_rvf[k];
    sh_flags[0] = (a > 0.0f) ? 1.0f : 0.0f;
  }
  for (int idx = tid; idx < KNB * 32; idx += 256) {
    int k = idx >> 5, f = idx & 31;
    float v;
    if      (f < 6)   v = sh_shifts[k][f];
    else if (f < 12)  v = (A.rsm[((size_t)b * KNB + k) * SS + (f - 6)] != 0) ? 1.0f : 0.0f;
    else if (f < 28)  v = A.emb16[sh_rrc[k] * 16 + (f - 12)];
    else if (f == 28) v = sh_rdv[k];
    else if (f == 29) v = sh_same[k];
    else              v = 0.0f;                       // pad 30 -> 32
    ov.a.nib[k * 36 + f] = f2bf(v);
  }
  for (int idx = tid; idx < HIDD * 32; idx += 256) {
    int kk = idx / HIDD, n = idx % HIDD;
    float w = (kk < 30) ? A.Wn[(size_t)kk * HIDD + n] : 0.0f;
    ov.a.wnT[n * 36 + kk] = f2bf(w);
  }
  __syncthreads();

  // ---- stage 2: WMMA kv = gelu(ni@Wn + bn), keep-mask -----------------------
  if (tid < KNB)
    sh_keep[tid] = (sh_rvf[tid] > 0.0f || (sh_flags[0] == 0.0f && tid == 0)) ? 1.0f : 0.0f;
#pragma unroll
  for (int i = 0; i < 6; ++i) {
    int t48 = wave * 6 + i;
    int mi = t48 & 3, nj = t48 >> 2;                  // 4 M-tiles x 12 N-tiles
    v16bf a  = load_a_frag(ov.a.nib, 36, mi * 16, 0, lane);
    v16bf bm = load_b_frag(ov.a.wnT, 36, nj * 16, 0, lane);
    v8f c = {};
    c = __builtin_amdgcn_wmma_f32_16x16x32_bf16(false, a, false, bm, (short)0, c, false, false);
    int n = nj * 16 + r16;
    float bias = A.bn[n];
#pragma unroll
    for (int ii = 0; ii < 8; ++ii) {
      int m = mi * 16 + half * 8 + ii;
      sh_kv[m * 200 + n] = f2bf(gelu_f(c[ii] + bias));
    }
  }
  __syncthreads();

  // ---- stage 3: fetch q/tq rows; neighbor statistics ----------------------
  if (tid < HIDD) {
    ov.bb.qv[tid]  = A.qrow[(size_t)b * HIDD + tid];
    ov.bb.tqv[tid] = A.tqrow[(size_t)b * HIDD + tid];
  } else if (tid < 200) {
    ov.bb.tracc[tid - 192] = 0.0f;
  } else if (tid >= 208 && tid < 208 + SS) {          // per-shift stats (pss)
    int s = tid - 208;
    float cnt = 0.0f, mean = 0.0f;
    for (int k = 0; k < KNB; ++k) { float v = sh_vmf[k][s]; cnt += v; mean += sh_shifts[k][s] * v; }
    float den = fmaxf(cnt, 1.0f); mean /= den;
    float var = 0.0f;
    for (int k = 0; k < KNB; ++k) { float d = sh_shifts[k][s] - mean; var += d * d * sh_vmf[k][s]; }
    var = fmaxf(var / den, 0.0f);
    ov.bb.pss[s][0] = cnt / (float)KNB;
    ov.bb.pss[s][1] = (cnt > 0.0f) ? 1.0f : 0.0f;
    ov.bb.pss[s][2] = mean;
    ov.bb.pss[s][3] = sqrtf(var);
  } else if (tid == 224) {                            // global stats (gst)
    float nv = 0.0f, srd = 0.0f, mx = -1e30f, ssm = 0.0f;
    for (int k = 0; k < KNB; ++k) {
      float vf = sh_rvf[k], rdv = sh_rdv[k] * vf;
      nv += vf; srd += rdv; mx = fmaxf(mx, rdv); ssm += sh_same[k] * vf;
    }
    float dn = fmaxf(nv, 1.0f);
    ov.bb.gst[0] = nv / (float)KNB; ov.bb.gst[1] = srd / dn;
    ov.bb.gst[2] = mx;              ov.bb.gst[3] = ssm / dn;
  }
  __syncthreads();

  // ---- stage 4: Q = q@Wqa+bqa ; shift-similarity bias sb ------------------
  if (tid < HIDD) {
    float acc = A.bqa[tid];
    for (int j = 0; j < HIDD; ++j) acc += ov.bb.qv[j] * A.Wqa[(size_t)j * HIDD + tid];
    ov.bb.Qv[tid] = acc;
  }
  {
    int k = tid & 63, slice = tid >> 6;               // 4 j-slices of 24 per k
    float sd[12];
#pragma unroll
    for (int s = 0; s < SS; ++s) {
      sd[s]     = (sh_qos[s] - sh_shifts[k][s]) * sh_qsm[s];
      sd[s + 6] = sh_qsm[s];
    }
    float part[NHEAD] = {0.f, 0.f, 0.f, 0.f};
    for (int jj = 0; jj < 24; ++jj) {
      int j = slice * 24 + jj;
      float acc = A.bs1[j];
#pragma unroll
      for (int i = 0; i < 12; ++i) acc += sd[i] * A.Ws1[i * 96 + j];
      float h1 = gelu_f(acc);
#pragma unroll
      for (int h = 0; h < NHEAD; ++h) part[h] += h1 * A.Ws2[j * NHEAD + h];
    }
#pragma unroll
    for (int h = 0; h < NHEAD; ++h) atomicAdd(&sh_sb[k][h], part[h]);
  }
  __syncthreads();

  // ---- stage 5: A[h,j] = sum_d Q[h,d]*Wka[j, h*48+d] (bka is softmax-inv.)
#pragma unroll
  for (int rep = 0; rep < 3; ++rep) {
    int idx = tid + rep * 256;                        // 768 = 4*192
    int h = idx / HIDD, j = idx % HIDD;
    float acc = 0.0f;
    for (int d = 0; d < HDIM; ++d)
      acc += ov.bb.Qv[h * HDIM + d] * A.Wka[(size_t)j * HIDD + h * HDIM + d];
    ov.bb.Aproj[h][j] = acc;
  }
  __syncthreads();

  // ---- stage 6: scores[h,k] = A[h]·kv[k]/sqrt(48) + sb + bs2, masked ------
  {
    int h = tid >> 6, k = tid & 63;
    float acc = 0.0f;
    for (int j = 0; j < HIDD; ++j) acc += ov.bb.Aproj[h][j] * bf2f(sh_kv[k * 200 + j]);
    float sc = acc * 0.14433756729740643f + sh_sb[k][h] + A.bs2[h];
    sh_scores[h][k] = (sh_keep[k] > 0.0f) ? sc : -10000.0f;
  }
  __syncthreads();

  // ---- stage 7: softmax over k per head -----------------------------------
  if (tid < NHEAD) {
    int h = tid;
    float mx = -3.4e38f;
    for (int k = 0; k < KNB; ++k) mx = fmaxf(mx, sh_scores[h][k]);
    float sum = 0.0f;
    for (int k = 0; k < KNB; ++k) { float e = expf(sh_scores[h][k] - mx); sh_scores[h][k] = e; sum += e; }
    float inv = 1.0f / sum;
    for (int k = 0; k < KNB; ++k) sh_scores[h][k] *= inv;
  }
  __syncthreads();

  // ---- stage 8: u[h,j] = sum_k attn[h,k]*kv[k,j] --------------------------
#pragma unroll
  for (int rep = 0; rep < 3; ++rep) {
    int idx = tid + rep * 256;
    int h = idx / HIDD, j = idx % HIDD;
    float acc = 0.0f;
    for (int k = 0; k < KNB; ++k) acc += sh_scores[h][k] * bf2f(sh_kv[k * 200 + j]);
    ov.bb.u[h][j] = acc;
  }
  __syncthreads();

  // ---- stage 9: ao[o] = sum_j u[h,j]*Wva[j,o] + bva[o]  (h = o/48) --------
  if (tid < HIDD) {
    int o = tid, h = o / HDIM;
    float acc = A.bva[o];
    for (int j = 0; j < HIDD; ++j) acc += ov.bb.u[h][j] * A.Wva[(size_t)j * HIDD + o];
    ov.bb.ao[o] = acc;
  }
  __syncthreads();

  // ---- stage 10: c1 = gelu(ao@Wc1+bc1), t1 = gelu(ao@Wt1+bt1) -------------
  for (int idx = tid; idx < 2 * HIDD; idx += 256) {
    int o = idx % HIDD;
    const float* W = (idx < HIDD) ? A.Wc1 : A.Wt1;
    const float* bi = (idx < HIDD) ? A.bc1 : A.bt1;
    float acc = bi[o];
    for (int j = 0; j < HIDD; ++j) acc += ov.bb.ao[j] * W[(size_t)j * HIDD + o];
    float* dst = (idx < HIDD) ? ov.bb.c1 : ov.bb.t1;
    dst[o] = gelu_f(acc);
  }
  __syncthreads();

  // ---- stage 11: ctx out, swb = t1@Wt2+bt2 --------------------------------
  const bool anyv = (sh_flags[0] > 0.0f);
  if (tid < HIDD) {
    int o = tid;
    float acc = A.bc2[o];
    for (int j = 0; j < HIDD; ++j) acc += ov.bb.c1[j] * A.Wc2[(size_t)j * HIDD + o];
    A.out_ctx[(size_t)b * HIDD + o] = anyv ? acc : A.fb_ctx[o];
  } else if (tid < HIDD + SS) {
    int s = tid - HIDD;
    float acc = A.bt2[s];
    for (int j = 0; j < HIDD; ++j) acc += ov.bb.t1[j] * A.Wt2[j * SS + s];
    ov.bb.swb[s] = acc;
  }
  __syncthreads();

  // ---- stage 12: transferred shifts; tc = gelu([tq,e32,gst]@Wtc+btc) ------
  if (tid < HIDD) {
    int o = tid;
    float acc = A.btc[o];
    for (int i = 0; i < HIDD; ++i) acc += ov.bb.tqv[i] * A.Wtc[(size_t)i * HIDD + o];
    for (int i = 0; i < 32; ++i)   acc += A.emb32[qrc_raw * 32 + i] * A.Wtc[(size_t)(HIDD + i) * HIDD + o];
    for (int i = 0; i < 4; ++i)    acc += ov.bb.gst[i] * A.Wtc[(size_t)(HIDD + 32 + i) * HIDD + o];
    ov.bb.tcv[o] = gelu_f(acc);
  }
  if (tid < SS) {
    int s = tid;
    float swbs = ov.bb.swb[s] * 0.1f;
    float mx = -3.4e38f;
    for (int k = 0; k < KNB; ++k) {
      float wv = (sh_vmf[k][s] > 0.0f) ? (sh_rdv[k] + sh_same[k] * 0.5f + swbs) : -10000.0f;
      mx = fmaxf(mx, wv);
    }
    float sum = 0.0f, wsum = 0.0f;
    for (int k = 0; k < KNB; ++k) {
      float wv = (sh_vmf[k][s] > 0.0f) ? (sh_rdv[k] + sh_same[k] * 0.5f + swbs) : -10000.0f;
      float e = expf(wv - mx);
      sum += e; wsum += e * sh_shifts[k][s];
    }
    float tr = (wsum / sum) * A.scale[s];
    A.out_trans[(size_t)b * SS + s] = anyv ? tr : A.fb_shift[s];
  }
  __syncthreads();

  // ---- stage 13: trust hidden layer ([tc,pss_s] @ Wp1, dot Wp2) -----------
  for (int idx = tid; idx < SS * 96; idx += 256) {
    int s = idx / 96, j = idx % 96;
    float acc = A.bp1[j];
    for (int i = 0; i < HIDD; ++i) acc += ov.bb.tcv[i] * A.Wp1[(size_t)i * 96 + j];
#pragma unroll
    for (int i = 0; i < 4; ++i) acc += ov.bb.pss[s][i] * A.Wp1[(size_t)(HIDD + i) * 96 + j];
    atomicAdd(&ov.bb.tracc[s], gelu_f(acc) * A.Wp2[j]);
  }
  __syncthreads();

  // ---- stage 14: trust out ------------------------------------------------
  if (tid < SS)
    A.out_trust[(size_t)b * SS + tid] = sigm(ov.bb.tracc[tid] + A.bp2[0]);
}

// ============================================================================
extern "C" void kernel_launch(void* const* d_in, const int* in_sizes, int n_in,
                              void* d_out, int out_size, void* d_ws, size_t ws_size,
                              hipStream_t stream) {
  (void)in_sizes; (void)n_in; (void)out_size; (void)ws_size;
  // inputs: setup_inputs() dict order; params flattened in insertion order
  const float*         qe    = (const float*)d_in[0];
  const int*           qrc   = (const int*)d_in[1];
  const float*         qos   = (const float*)d_in[2];
  const float*         qsm   = (const float*)d_in[3];
  const float*         rs    = (const float*)d_in[4];
  const unsigned char* rsm   = (const unsigned char*)d_in[5];
  const int*           rrc   = (const int*)d_in[6];
  const float*         rd    = (const float*)d_in[7];
  const unsigned char* rv    = (const unsigned char*)d_in[8];
  const float*         rctab = (const float*)d_in[9];
  const float* Wq1 = (const float*)d_in[10]; const float* bq1 = (const float*)d_in[11];
  const float* emb16 = (const float*)d_in[12];
  const float* Wn  = (const float*)d_in[13]; const float* bn  = (const float*)d_in[14];
  const float* Wqa = (const float*)d_in[15]; const float* bqa = (const float*)d_in[16];
  const float* Wka = (const float*)d_in[17]; /* bka d_in[18]: softmax-invariant */
  const float* Wva = (const float*)d_in[19]; const float* bva = (const float*)d_in[20];
  const float* Ws1 = (const float*)d_in[21]; const float* bs1 = (const float*)d_in[22];
  const float* Ws2 = (const float*)d_in[23]; const float* bs2 = (const float*)d_in[24];
  const float* Wc1 = (const float*)d_in[25]; const float* bc1 = (const float*)d_in[26];
  const float* Wc2 = (const float*)d_in[27]; const float* bc2 = (const float*)d_in[28];
  const float* Wt1 = (const float*)d_in[29]; const float* bt1 = (const float*)d_in[30];
  const float* Wt2 = (const float*)d_in[31]; const float* bt2 = (const float*)d_in[32];
  const float* emb32 = (const float*)d_in[33];
  const float* Wtq = (const float*)d_in[34]; const float* btq = (const float*)d_in[35];
  const float* Wtc = (const float*)d_in[36]; const float* btc = (const float*)d_in[37];
  const float* Wp1 = (const float*)d_in[38]; const float* bp1 = (const float*)d_in[39];
  const float* Wp2 = (const float*)d_in[40]; const float* bp2 = (const float*)d_in[41];
  const float* scale    = (const float*)d_in[42];
  const float* fb_shift = (const float*)d_in[43];
  const float* fb_ctx   = (const float*)d_in[44];

  float* wsf   = (float*)d_ws;
  float* qbuf  = wsf;
  float* tqbuf = wsf + (size_t)BB * HIDD;

  qtq_gemm_kernel<<<BB / 32, 256, 0, stream>>>(qe, Wq1, bq1, Wtq, btq, qbuf, tqbuf);

  FusedArgs a;
  a.qe = qe; a.qrc = qrc; a.qos = qos; a.qsm = qsm;
  a.rs = rs; a.rsm = rsm; a.rrc = rrc; a.rd = rd; a.rv = rv; a.rctab = rctab;
  a.emb16 = emb16; a.Wn = Wn; a.bn = bn; a.Wqa = Wqa; a.bqa = bqa;
  a.Wka = Wka; a.Wva = Wva; a.bva = bva;
  a.Ws1 = Ws1; a.bs1 = bs1; a.Ws2 = Ws2; a.bs2 = bs2;
  a.Wc1 = Wc1; a.bc1 = bc1; a.Wc2 = Wc2; a.bc2 = bc2;
  a.Wt1 = Wt1; a.bt1 = bt1; a.Wt2 = Wt2; a.bt2 = bt2;
  a.emb32 = emb32; a.Wtc = Wtc; a.btc = btc;
  a.Wp1 = Wp1; a.bp1 = bp1; a.Wp2 = Wp2; a.bp2 = bp2;
  a.scale = scale; a.fb_shift = fb_shift; a.fb_ctx = fb_ctx;
  a.qrow = qbuf; a.tqrow = tqbuf;
  a.out_ctx   = (float*)d_out;
  a.out_trans = (float*)d_out + (size_t)BB * HIDD;
  a.out_trust = (float*)d_out + (size_t)BB * HIDD + (size_t)BB * SS;

  fused_kernel<<<BB, 256, 0, stream>>>(a);
}
